// NeuralLRURec_50878182588529
// MI455X (gfx1250) — matrix operand order repaired and checked by reading, not compiled, not run on hardware
//
#include <hip/hip_runtime.h>
#include <hip/hip_bf16.h>
#include <math.h>

typedef _Float16 h8_t  __attribute__((ext_vector_type(8)));
typedef _Float16 h16_t __attribute__((ext_vector_type(16)));
typedef float    f8_t  __attribute__((ext_vector_type(8)));
typedef int      i4_t  __attribute__((ext_vector_type(4)));

#define NBATCH 8
#define LSEQ   512
#define MROWS  (NBATCH*LSEQ)   // 4096
#define DM     256
#define DD     512
#define DFF    1024
#define VV     32001
#define VPAD   32064           // round up to multiple of 64

// ---- async global->LDS path (CDNA5); guarded so either toolchain compiles ----
#if __has_builtin(__builtin_amdgcn_global_load_async_to_lds_b128) && \
    __has_builtin(__builtin_amdgcn_s_wait_asynccnt)
#define ASYNC_LDS 1
#define AWAIT(n)  __builtin_amdgcn_s_wait_asynccnt(n)
#else
#define ASYNC_LDS 0
#define AWAIT(n)
#endif
// builtin takes (v4i addrspace(1)*, v4i addrspace(3)*, imm offset, imm cpol)
#define ASG(p) ((__attribute__((address_space(1))) i4_t*)(p))
#define ASL(p) ((__attribute__((address_space(3))) i4_t*)(p))

// ---------------- workspace layout (bytes) ----------------
constexpr size_t OFF_X   = 0;                                  // f32 [4096,256]
constexpr size_t SZ_X    = (size_t)MROWS*DM*4;
constexpr size_t OFF_XH  = OFF_X + SZ_X;                       // f16 packed-A [4096,256]
constexpr size_t SZ_XH   = (size_t)MROWS*DM*2;
constexpr size_t OFF_T1  = OFF_XH + SZ_XH;                     // f32 [4096,1024]
constexpr size_t SZ_T1   = (size_t)MROWS*DFF*4;
constexpr size_t OFF_T1H = OFF_T1 + SZ_T1;                     // f16 packed-A [4096,1024]
constexpr size_t SZ_T1H  = (size_t)MROWS*DFF*2;
constexpr size_t OFF_T2  = OFF_T1H + SZ_T1H;                   // f32 [4096,256]
constexpr size_t SZ_T2   = (size_t)MROWS*DM*4;
constexpr size_t OFF_WT  = OFF_T2 + SZ_T2;                     // f16 [K<=1024, Npad<=32064]
constexpr size_t SZ_WT   = (size_t)DM*VPAD*2;
constexpr size_t OFF_LRE = OFF_WT + SZ_WT;                     // f32 [512] x3
constexpr size_t OFF_LIM = OFF_LRE + DD*4;
constexpr size_t OFF_GAM = OFF_LIM + DD*4;

// ================= embedding lookup + layernorm =================
__global__ __launch_bounds__(256) void k_embed_ln(const int* __restrict__ tok,
        const float* __restrict__ embed, const float* __restrict__ w,
        const float* __restrict__ b, float* __restrict__ x)
{
  __shared__ float sm[256];
  const int r = blockIdx.x, c = threadIdx.x;
  const int t = tok[r];
  float v = embed[(size_t)t*DM + c];
  sm[c] = v; __syncthreads();
  #pragma unroll
  for (int s = 128; s > 0; s >>= 1) { if (c < s) sm[c] += sm[c+s]; __syncthreads(); }
  float mu = sm[0] * (1.f/DM); __syncthreads();
  float d = v - mu; sm[c] = d*d; __syncthreads();
  #pragma unroll
  for (int s = 128; s > 0; s >>= 1) { if (c < s) sm[c] += sm[c+s]; __syncthreads(); }
  float var = sm[0] * (1.f/DM);
  x[(size_t)r*DM + c] = d * rsqrtf(var + 1e-5f) * w[c] + b[c];
}

// ============ row layernorm with fused (gemm_out + bias + residual) =========
// biasc stride 2 -> real part of a complex64 bias; stride 1 -> plain f32 bias
__global__ __launch_bounds__(256) void k_row_ln(const float* __restrict__ g,
        const float* __restrict__ biasc, int bstride,
        float* __restrict__ x, const float* __restrict__ w, const float* __restrict__ b)
{
  __shared__ float sm[256];
  const int r = blockIdx.x, c = threadIdx.x;
  const size_t i = (size_t)r*DM + c;
  float v = g[i] + biasc[(size_t)c*bstride] + x[i];
  sm[c] = v; __syncthreads();
  #pragma unroll
  for (int s = 128; s > 0; s >>= 1) { if (c < s) sm[c] += sm[c+s]; __syncthreads(); }
  float mu = sm[0] * (1.f/DM); __syncthreads();
  float d = v - mu; sm[c] = d*d; __syncthreads();
  #pragma unroll
  for (int s = 128; s > 0; s >>= 1) { if (c < s) sm[c] += sm[c+s]; __syncthreads(); }
  float var = sm[0] * (1.f/DM);
  x[i] = d * rsqrtf(var + 1e-5f) * w[c] + b[c];
}

// ================= per-channel lambda / gamma =================
__global__ void k_lambda(const float* __restrict__ pl, float* __restrict__ lre,
                         float* __restrict__ lim, float* __restrict__ gam)
{
  int d = blockIdx.x * blockDim.x + threadIdx.x;
  if (d >= DD) return;
  float nu = expf(pl[d]);
  float th = expf(pl[DD + d]);
  float mag = expf(-nu);
  lre[d] = mag * cosf(th);
  lim[d] = mag * sinf(th);
  gam[d] = expf(pl[2*DD + d]);
}

// ============ h = (gemm_out + b_in) * gamma, packed [re | im] along N ========
__global__ void k_hpost(float* __restrict__ t1, const float* __restrict__ binc,
                        const float* __restrict__ gam)
{
  int i = blockIdx.x*256 + threadIdx.x;         // over 4096*1024
  int dd = i & (DFF-1); int d = dd & (DD-1);
  float b = binc[2*d + (dd >= DD ? 1 : 0)];
  t1[i] = (t1[i] + b) * gam[d];
}

// ================= masked LRU scan: h[t] = x[t] + m[t-1]*lamb*h[t-1] ========
__global__ void k_scan(float* __restrict__ t1, const float* __restrict__ lre,
                       const float* __restrict__ lim, const int* __restrict__ tok)
{
  int idx = blockIdx.x * blockDim.x + threadIdx.x;   // 0..4095
  int b = idx >> 9, d = idx & (DD-1);
  const float ar = lre[d], ai = lim[d];
  const int* tb = tok + (size_t)b*LSEQ;
  float* base = t1 + (size_t)b*LSEQ*DFF + d;
  float hr = 0.f, hi = 0.f;
  int pm = 0;
  for (int t = 0; t < LSEQ; ++t) {
    if (t + 8 < LSEQ) {
      __builtin_prefetch(base + (size_t)(t+8)*DFF, 0, 1);
      __builtin_prefetch(base + (size_t)(t+8)*DFF + DD, 0, 1);
    }
    float xr = base[(size_t)t*DFF];
    float xi = base[(size_t)t*DFF + DD];
    if (pm) {
      float nr = xr + ar*hr - ai*hi;
      float ni = xi + ar*hi + ai*hr;
      hr = nr; hi = ni;
    } else { hr = xr; hi = xi; }
    base[(size_t)t*DFF]      = hr;
    base[(size_t)t*DFF + DD] = hi;
    pm = tb[t] > 0;
  }
}

// ================= GELU (exact) with fused bias =================
__global__ void k_gelu(float* __restrict__ t1, const float* __restrict__ b1)
{
  int i = blockIdx.x*256 + threadIdx.x;          // over 4096*1024
  int c = i & (DFF-1);
  float v = t1[i] + b1[c];
  t1[i] = 0.5f * v * (1.f + erff(v * 0.70710678118f));
}

// ====== fragment-ready A packing: f32 [M,K] -> f16 tiles ====================
// Tile (rt: 16 rows, kt: 32 K). Within a tile, lane l owns 16 contiguous f16:
//   m = rt*16 + (l&15), hi = l>>4,
//   element e -> k = kt*32 + (e<8 ? hi*8+e : 16 + hi*8 + (e-8))
__global__ void k_pack_a(const float* __restrict__ s, _Float16* __restrict__ d, int K)
{
  int i = blockIdx.x*256 + threadIdx.x;          // over M*K
  int e    = i & 15;
  int l    = (i >> 4) & 31;
  int tile = i >> 9;
  int ktiles = K >> 5;
  int kt = tile % ktiles;
  int rt = tile / ktiles;
  int m  = rt*16 + (l & 15);
  int hi = l >> 4;
  int k  = kt*32 + ((e < 8) ? (hi*8 + e) : (16 + hi*8 + (e - 8)));
  d[i] = (_Float16)s[(size_t)m*K + k];
}

// ================= weight packing =================
// W [N,K] f32 row-major  ->  Wt [K, Npad] f16  (zero-pad n>=N)
__global__ void k_pack_wt(const float* __restrict__ W, _Float16* __restrict__ Wt,
                          int N, int K, int Npad)
{
  int i = blockIdx.x*256 + threadIdx.x;
  if (i >= K*Npad) return;
  int k = i / Npad, n = i % Npad;
  Wt[i] = (n < N) ? (_Float16)W[(size_t)n*K + k] : (_Float16)0.f;
}
// W_in complex64 [512,256] -> Wt [256, 1024] f16 : cols [Re(512) | Im(512)]
__global__ void k_pack_win(const float* __restrict__ Wc, _Float16* __restrict__ Wt)
{
  int i = blockIdx.x*256 + threadIdx.x;          // over 256*1024
  int k = i >> 10, n = i & 1023;
  float v = (n < DD) ? Wc[((size_t)n*DM + k)*2]
                     : Wc[((size_t)(n-DD)*DM + k)*2 + 1];
  Wt[i] = (_Float16)v;
}
// W_out complex64 [256,512] -> Wt [1024, 256] f16 : rows [Re(512) ; -Im(512)]
__global__ void k_pack_wout(const float* __restrict__ Wc, _Float16* __restrict__ Wt)
{
  int i = blockIdx.x*256 + threadIdx.x;          // over 1024*256
  int k = i >> 8, n = i & 255;
  float v = (k < DD) ?  Wc[((size_t)n*DD + k)*2]
                     : -Wc[((size_t)n*DD + (k-DD))*2 + 1];
  Wt[i] = (_Float16)v;
}

// ================= WMMA GEMM:  C[M,Nld] = A[M,K] * Wt[K,Npad]  (+ biasN) =====
// 128 threads = 4 wave32; wave w -> 16-row strip; block tile 64x64.
// B tile (64K x 64N) is double-buffered in LDS, filled with async global->LDS
// DMA (ASYNCcnt) and shared by all 4 waves (4x less global B traffic).
// Rows padded to 72 f16 (144B) so the 16B chunk writes from fill stay 16B-aligned.
#define LDSROW 72
#define STAGE  (64*LDSROW)

__global__ __launch_bounds__(128) void k_gemm(const _Float16* __restrict__ A,
        const _Float16* __restrict__ Wt, float* __restrict__ C,
        const float* __restrict__ biasN, int M, int K, int N, int Npad, int Nld)
{
  __shared__ _Float16 lb[2*STAGE];
  const int tid  = threadIdx.x;
  const int lane = tid & 31;
  const int wave = tid >> 5;
  const int mbase = blockIdx.y*64 + wave*16;
  const int nbase = blockIdx.x*64;
  const int hi = lane >> 4;
  const int mr = lane & 15;

  f8_t acc[4] = {};

  // packed-A: strip rt = mbase/16 holds K/32 tiles of 512 f16; lane owns 16 each
  const _Float16* aptr = A + ((size_t)(mbase >> 4) * (K >> 5) * 32 + lane) * 16;
  const _Float16* gB   = Wt + nbase;
  const int nkb = K >> 6;                          // 64-wide K blocks

  auto fill = [&](int stage, int kb) {             // 512 16B chunks / 128 threads
    const _Float16* src = gB + ((size_t)kb << 6) * Npad;
    _Float16* dst = lb + stage*STAGE;
    #pragma unroll
    for (int j = 0; j < 4; ++j) {
      int c   = tid + 128*j;
      int row = c >> 3, ck = c & 7;
      const _Float16* g = src + (size_t)row*Npad + ck*8;
      _Float16*       l = dst + row*LDSROW + ck*8;
#if ASYNC_LDS
      __builtin_amdgcn_global_load_async_to_lds_b128(ASG(g), ASL(l), 0, 0);
#else
      *(h8_t*)l = *(const h8_t*)g;
#endif
    }
  };

  auto compute = [&](int kb, int st) {
    h16_t a0 = *(const h16_t*)(aptr + (size_t)(2*kb)   * 512);
    h16_t a1 = *(const h16_t*)(aptr + (size_t)(2*kb+1) * 512);
    const _Float16* l0 = lb + st*STAGE + mr;               // K rows 0..31 of block
    const _Float16* l1 = l0 + 32*LDSROW;                   // K rows 32..63
    #pragma unroll
    for (int t = 0; t < 4; ++t) {
      h16_t b0;
      #pragma unroll
      for (int s = 0; s < 16; ++s)
        b0[s] = l0[((s < 8) ? (hi*8 + s) : (8 + hi*8 + s))*LDSROW + t*16];
      acc[t] = __builtin_amdgcn_wmma_f32_16x16x32_f16(
                   false, a0, false, b0, (short)0, acc[t], false, false);
    }
    #pragma unroll
    for (int t = 0; t < 4; ++t) {
      h16_t b1;
      #pragma unroll
      for (int s = 0; s < 16; ++s)
        b1[s] = l1[((s < 8) ? (hi*8 + s) : (8 + hi*8 + s))*LDSROW + t*16];
      acc[t] = __builtin_amdgcn_wmma_f32_16x16x32_f16(
                   false, a1, false, b1, (short)0, acc[t], false, false);
    }
  };

  fill(0, 0);
  int s = 0;
  for (int kb = 0; kb < nkb - 1; ++kb) {
    fill(s ^ 1, kb + 1);        // next stage DMA in flight under this stage's WMMAs
    AWAIT(4);                   // our 4 earlier ops (current stage) are complete
    __syncthreads();            // every wave's part of current stage visible
    compute(kb, s);
    __syncthreads();            // stage fully consumed before it is refilled
    s ^= 1;
  }
  AWAIT(0);
  __syncthreads();
  compute(nkb - 1, s);

  // D layout: vgpr r -> M = mbase + 8*hi + r, N = nbase + 16*t + (lane&15)
  #pragma unroll
  for (int t = 0; t < 4; ++t) {
    int n = nbase + t*16 + mr;
    if (n < N) {
      float bv = biasN ? biasN[n] : 0.f;
      #pragma unroll
      for (int r = 0; r < 8; ++r) {
        int m = mbase + hi*8 + r;
        C[(size_t)m*Nld + n] = acc[t][r] + bv;
      }
    }
  }
}

// =============================== host ===============================
extern "C" void kernel_launch(void* const* d_in, const int* in_sizes, int n_in,
                              void* d_out, int out_size, void* d_ws, size_t ws_size,
                              hipStream_t stream)
{
  (void)in_sizes; (void)n_in; (void)out_size; (void)ws_size;
  const int*   tok   = (const int*)  d_in[0];
  const float* embed = (const float*)d_in[1];
  const float* elnw  = (const float*)d_in[2];
  const float* elnb  = (const float*)d_in[3];
  const float* bias  = (const float*)d_in[30];

  char* ws = (char*)d_ws;
  float*    x   = (float*)   (ws + OFF_X);
  _Float16* xh  = (_Float16*)(ws + OFF_XH);
  float*    t1  = (float*)   (ws + OFF_T1);
  _Float16* t1h = (_Float16*)(ws + OFF_T1H);
  float*    t2  = (float*)   (ws + OFF_T2);
  _Float16* Wt  = (_Float16*)(ws + OFF_WT);
  float*    lre = (float*)   (ws + OFF_LRE);
  float*    lim = (float*)   (ws + OFF_LIM);
  float*    gam = (float*)   (ws + OFF_GAM);

  // embed + layernorm -> x [4096,256]
  k_embed_ln<<<MROWS, 256, 0, stream>>>(tok, embed, elnw, elnb, x);

  for (int blk = 0; blk < 2; ++blk) {
    const int base = 4 + blk*13;
    const float* pl   = (const float*)d_in[base+0];
    const float* Win  = (const float*)d_in[base+1];   // complex interleaved
    const float* bin  = (const float*)d_in[base+2];   // complex interleaved
    const float* Wout = (const float*)d_in[base+3];   // complex interleaved
    const float* bout = (const float*)d_in[base+4];   // complex interleaved
    const float* ln1w = (const float*)d_in[base+5];
    const float* ln1b = (const float*)d_in[base+6];
    const float* W1   = (const float*)d_in[base+7];
    const float* b1   = (const float*)d_in[base+8];
    const float* W2   = (const float*)d_in[base+9];
    const float* b2   = (const float*)d_in[base+10];
    const float* ln2w = (const float*)d_in[base+11];
    const float* ln2b = (const float*)d_in[base+12];

    k_lambda<<<2, 256, 0, stream>>>(pl, lre, lim, gam);

    // ---- input projection: h = (x @ W_in^T + b_in) * gamma, packed [re|im] ----
    k_pack_a<<<(MROWS*DM)/256, 256, 0, stream>>>(x, xh, DM);
    k_pack_win<<<(DM*DFF)/256, 256, 0, stream>>>(Win, Wt);
    k_gemm<<<dim3(DFF/64, MROWS/64), 128, 0, stream>>>(xh, Wt, t1, nullptr,
                                                       MROWS, DM, DFF, DFF, DFF);
    k_hpost<<<(MROWS*DFF)/256, 256, 0, stream>>>(t1, bin, gam);

    // ---- masked LRU scan over L, parallel over B*D channels ----
    k_scan<<<(NBATCH*DD)/256, 256, 0, stream>>>(t1, lre, lim, tok);

    // ---- output projection: Re(h @ W_out^T) = hr@Re^T - hi@Im^T ----
    k_pack_a<<<(MROWS*DFF)/256, 256, 0, stream>>>(t1, t1h, DFF);
    k_pack_wout<<<(DFF*DM)/256, 256, 0, stream>>>(Wout, Wt);
    k_gemm<<<dim3(DM/64, MROWS/64), 128, 0, stream>>>(t1h, Wt, t2, nullptr,
                                                      MROWS, DFF, DM, DM, DM);
    k_row_ln<<<MROWS, 256, 0, stream>>>(t2, bout, 2, x, ln1w, ln1b);

    // ---- FFN: gelu(x@W1^T + b1) @ W2^T + b2 + x, layernorm ----
    k_pack_a<<<(MROWS*DM)/256, 256, 0, stream>>>(x, xh, DM);
    k_pack_wt<<<(DM*DFF)/256, 256, 0, stream>>>(W1, Wt, DFF, DM, DFF);
    k_gemm<<<dim3(DFF/64, MROWS/64), 128, 0, stream>>>(xh, Wt, t1, nullptr,
                                                       MROWS, DM, DFF, DFF, DFF);
    k_gelu<<<(MROWS*DFF)/256, 256, 0, stream>>>(t1, b1);
    k_pack_a<<<(MROWS*DFF)/256, 256, 0, stream>>>(t1, t1h, DFF);
    k_pack_wt<<<(DFF*DM)/256, 256, 0, stream>>>(W2, Wt, DM, DFF, DM);
    k_gemm<<<dim3(DM/64, MROWS/64), 128, 0, stream>>>(t1h, Wt, t2, nullptr,
                                                      MROWS, DFF, DM, DM, DM);
    k_row_ln<<<MROWS, 256, 0, stream>>>(t2, b2, 1, x, ln2w, ln2b);
  }

  // ---- vocab projection: logits = x @ embed^T + bias (bias fused in store) ----
  k_pack_a<<<(MROWS*DM)/256, 256, 0, stream>>>(x, xh, DM);
  k_pack_wt<<<(DM*VPAD)/256, 256, 0, stream>>>(embed, Wt, VV, DM, VPAD);
  k_gemm<<<dim3(VPAD/64, MROWS/64), 128, 0, stream>>>(xh, Wt, (float*)d_out, bias,
                                                      MROWS, DM, VV, VPAD, VV);
}
